// lstm_27333171871751
// MI455X (gfx1250) — matrix operand order repaired and checked by reading, not compiled, not run on hardware
//
#include <hip/hip_runtime.h>
#include <hip/hip_bf16.h>

// ---------------------------------------------------------------------------
// Fused 2-layer LSTM (B=512, T=1024, IN=49, H=32, OUT=10) for gfx1250.
// 32 workgroups x 128 threads (4 waves). Each WG owns a 16-batch tile and
// runs the full time scan with WMMA f16 GEMMs from LDS-resident weights.
// x tiles staged with global_load_async_to_lds_b32 (ASYNCcnt-tracked);
// gate nonlinearities use v_tanh_f32 when available.
// ---------------------------------------------------------------------------

typedef __attribute__((ext_vector_type(16))) _Float16 v16h;
typedef __attribute__((ext_vector_type(8)))  _Float16 v8h;
typedef __attribute__((ext_vector_type(8)))  float    v8f;
typedef __attribute__((ext_vector_type(4)))  float    v4f;

#define LSTM_IN   49
#define LSTM_H    32
#define LSTM_G    128   // 4*H
#define LSTM_B    512
#define LSTM_T    1024
#define LSTM_OUT  10
#define BTILE     16

__device__ __forceinline__ v8f wmma_f16(v16h a, v16h b, v8f c) {
  // D = A(16x32 f16) * B(32x16 f16) + C(16x16 f32)
  return __builtin_amdgcn_wmma_f32_16x16x32_f16(
      /*neg_a=*/false, a, /*neg_b=*/false, b,
      /*c_mod=*/(short)0, c, /*reuse_a=*/false, /*reuse_b=*/false);
}

// Branch-free tanh: hardware v_tanh_f32 on gfx1250 if the builtin exists.
__device__ __forceinline__ float fast_tanh(float x) {
#if __has_builtin(__builtin_amdgcn_tanhf)
  return __builtin_amdgcn_tanhf(x);
#elif __has_builtin(__builtin_amdgcn_tanh_f32)
  return __builtin_amdgcn_tanh_f32(x);
#else
  float e = __expf(2.0f * x);          // inf-safe: large |x| -> +-1
  return 1.0f - 2.0f / (e + 1.0f);
#endif
}
__device__ __forceinline__ float fast_sigmoid(float x) {
  return __builtin_fmaf(0.5f, fast_tanh(0.5f * x), 0.5f);
}

// A fragment (16x32 f16) from row-major f16 LDS buffer (rowStride in halves).
// Lane layout (ISA 7.12.2): m = lane&15; K chunks {kb..kb+7, kb+16..kb+23},
// kb = 8*(lane>=16).
__device__ __forceinline__ v16h load_a_frag(const _Float16* base, int rowStride,
                                            int lane, int kOff) {
  int m  = lane & 15;
  int kb = ((lane >> 4) << 3) + kOff;
  const _Float16* p = base + m * rowStride + kb;
  v8h lo = *(const v8h*)(p);        // K = kb .. kb+7
  v8h hi = *(const v8h*)(p + 16);   // K = kb+16 .. kb+23
  v16h r;
#pragma unroll
  for (int i = 0; i < 8; ++i) { r[i] = lo[i]; r[i + 8] = hi[i]; }
  return r;
}

// Same A fragment, sourced from a row-major f32 LDS buffer with on-the-fly
// f32->f16 conversion (rowStride in floats).
__device__ __forceinline__ v16h load_a_frag_f32(const float* base, int rowStride,
                                                int lane, int kOff) {
  int m  = lane & 15;
  int kb = ((lane >> 4) << 3) + kOff;
  const float* p = base + m * rowStride + kb;
  v4f a0 = *(const v4f*)(p);
  v4f a1 = *(const v4f*)(p + 4);
  v4f b0 = *(const v4f*)(p + 16);
  v4f b1 = *(const v4f*)(p + 20);
  v16h r;
#pragma unroll
  for (int i = 0; i < 4; ++i) {
    r[i]      = (_Float16)a0[i];
    r[i + 4]  = (_Float16)a1[i];
    r[i + 8]  = (_Float16)b0[i];
    r[i + 12] = (_Float16)b1[i];
  }
  return r;
}

// B fragment (32x16 f16) from column-major LDS tile: element (n,k) at n*KT + k.
// Lane holds column n = lane&15, rows K chunks {kb..kb+7, kb+16..kb+23}, kb = kOff + 8*(lane>=16).
__device__ __forceinline__ v16h load_b_frag(const _Float16* tileBase, int KT,
                                            int lane, int kOff) {
  int n = lane & 15;
  int g = lane >> 4;
  const _Float16* p = tileBase + n * KT + kOff + (g << 3);
  v8h lo = *(const v8h*)(p);
  v8h hi = *(const v8h*)(p + 16);
  v16h r;
#pragma unroll
  for (int i = 0; i < 8; ++i) { r[i] = lo[i]; r[i + 8] = hi[i]; }
  return r;
}

struct LstmSmem {
  _Float16 wih0[8][16][64];   // W_ih0 tiles, col-major per tile, K padded 49->64
  _Float16 whh0[8][16][32];   // W_hh0 tiles, col-major per tile
  _Float16 wih1[8][16][32];   // W_ih1 tiles
  _Float16 whh1[8][16][32];   // W_hh1 tiles
  _Float16 wout16[16][32];    // W_out rows, N padded 10->16 (B tile for proj WMMA)
  float    bias0[LSTM_G];     // b_ih0 + b_hh0
  float    bias1[LSTM_G];     // b_ih1 + b_hh1
  float    bout16[16];        // b_out padded to 16
  float    xf32[BTILE][64];   // x_t tile (f32), K padded to 64; pad zeroed once
  _Float16 h0buf[BTILE][LSTM_H];
  _Float16 h1buf[BTILE][LSTM_H];
  float    gbuf[LSTM_G][BTILE];  // gate pre-activations, column-major [col][m]
};

__global__ __launch_bounds__(128)
void lstm2_fused_wmma(const float* __restrict__ x,
                      const float* __restrict__ W_ih0,
                      const float* __restrict__ W_hh0,
                      const float* __restrict__ b_ih0,
                      const float* __restrict__ b_hh0,
                      const float* __restrict__ W_ih1,
                      const float* __restrict__ W_hh1,
                      const float* __restrict__ b_ih1,
                      const float* __restrict__ b_hh1,
                      const float* __restrict__ W_out,
                      const float* __restrict__ b_out,
                      float* __restrict__ out) {
  __shared__ LstmSmem s;

  const int tid  = threadIdx.x;
  const int lane = tid & 31;
  const int wid  = tid >> 5;
  const int b0   = blockIdx.x * BTILE;

  // ---- Preload + convert weights into LDS (f16, B-matrix col-major tiles) ----
  for (int i = tid; i < 8 * 16 * 64; i += 128) {       // W_ih0 [128][49] -> pad 64
    int tile = i >> 10, n = (i >> 6) & 15, k = i & 63;
    int row = tile * 16 + n;
    float v = (k < LSTM_IN) ? W_ih0[row * LSTM_IN + k] : 0.0f;
    s.wih0[tile][n][k] = (_Float16)v;
  }
  for (int i = tid; i < 8 * 16 * 32; i += 128) {       // [128][32] matrices
    int tile = i >> 9, n = (i >> 5) & 15, k = i & 31;
    int row = tile * 16 + n;
    s.whh0[tile][n][k] = (_Float16)W_hh0[row * LSTM_H + k];
    s.wih1[tile][n][k] = (_Float16)W_ih1[row * LSTM_H + k];
    s.whh1[tile][n][k] = (_Float16)W_hh1[row * LSTM_H + k];
  }
  for (int i = tid; i < 16 * 32; i += 128) {           // W_out [10][32] -> pad 16
    int n = i >> 5, k = i & 31;
    float v = (n < LSTM_OUT) ? W_out[n * LSTM_H + k] : 0.0f;
    s.wout16[n][k] = (_Float16)v;
  }
  for (int i = tid; i < LSTM_G; i += 128) {
    s.bias0[i] = b_ih0[i] + b_hh0[i];
    s.bias1[i] = b_ih1[i] + b_hh1[i];
  }
  if (tid < 16) s.bout16[tid] = (tid < LSTM_OUT) ? b_out[tid] : 0.0f;
  // zero initial hidden states + x pad columns (written once, never touched again)
  for (int i = tid; i < BTILE * LSTM_H; i += 128) {
    s.h0buf[i >> 5][i & 31] = (_Float16)0.0f;
    s.h1buf[i >> 5][i & 31] = (_Float16)0.0f;
  }
  for (int i = tid; i < BTILE * (64 - LSTM_IN); i += 128) {
    int bl = i / (64 - LSTM_IN), k = LSTM_IN + i % (64 - LSTM_IN);
    s.xf32[bl][k] = 0.0f;
  }
  __syncthreads();

  // Per-thread persistent cell state: element (bb+j, hh)
  const int hh = tid & 31;
  const int bb = (tid >> 5) << 2;
  float c0[4] = {0.f, 0.f, 0.f, 0.f};
  float c1[4] = {0.f, 0.f, 0.f, 0.f};

  for (int t = 0; t < LSTM_T; ++t) {
    // ---- Stage x_t tile via async global->LDS copy (ASYNCcnt path) ----
    for (int i = tid; i < BTILE * LSTM_IN; i += 128) {
      int bl = i / LSTM_IN, k = i % LSTM_IN;
      const float* ga = &x[((size_t)(b0 + bl) * LSTM_T + t) * LSTM_IN + k];
      unsigned ldsa = (unsigned)(unsigned long long)&s.xf32[bl][k];
      asm volatile("global_load_async_to_lds_b32 %0, %1, off"
                   :: "v"(ldsa), "v"(ga) : "memory");
    }
    // Prefetch next timestep's rows (global_prefetch_b8 on gfx1250).
    if (t + 1 < LSTM_T && tid < BTILE) {
      __builtin_prefetch(&x[((size_t)(b0 + tid) * LSTM_T + (t + 1)) * LSTM_IN], 0, 3);
    }
    asm volatile("s_wait_asynccnt 0x0" ::: "memory");
    __syncthreads();  // (A)

    // ---- Layer 0 gates: x @ W_ih0^T + h0 @ W_hh0^T + bias  (2 N-tiles/wave) ----
    {
      v16h a_h = load_a_frag(&s.h0buf[0][0], LSTM_H, lane, 0);
      v16h ax0 = load_a_frag_f32(&s.xf32[0][0], 64, lane, 0);
      v16h ax1 = load_a_frag_f32(&s.xf32[0][0], 64, lane, 32);
#pragma unroll
      for (int u = 0; u < 2; ++u) {
        int nt = wid * 2 + u;
        float bv = s.bias0[nt * 16 + (lane & 15)];
        v8f acc = {bv, bv, bv, bv, bv, bv, bv, bv};
        acc = wmma_f16(ax0, load_b_frag(&s.wih0[nt][0][0], 64, lane, 0), acc);
        acc = wmma_f16(ax1, load_b_frag(&s.wih0[nt][0][0], 64, lane, 32), acc);
        acc = wmma_f16(a_h, load_b_frag(&s.whh0[nt][0][0], 32, lane, 0), acc);
        // D layout: M = v + 8*(lane>=16), N = lane&15 -> contiguous 8-float store
        *(v8f*)&s.gbuf[nt * 16 + (lane & 15)][(lane >> 4) << 3] = acc;
      }
    }
    __syncthreads();  // (B)

    // ---- Layer 0 elementwise (i,f,g,o -> c,h) ----
    {
      v4f vi = *(const v4f*)&s.gbuf[0 + hh][bb];
      v4f vf = *(const v4f*)&s.gbuf[32 + hh][bb];
      v4f vg = *(const v4f*)&s.gbuf[64 + hh][bb];
      v4f vo = *(const v4f*)&s.gbuf[96 + hh][bb];
#pragma unroll
      for (int j = 0; j < 4; ++j) {
        float ig = fast_sigmoid(vi[j]);
        float fg = fast_sigmoid(vf[j]);
        float gg = fast_tanh(vg[j]);
        float og = fast_sigmoid(vo[j]);
        c0[j] = fg * c0[j] + ig * gg;
        s.h0buf[bb + j][hh] = (_Float16)(og * fast_tanh(c0[j]));
      }
    }
    __syncthreads();  // (C)

    // ---- Layer 1 gates: h0 @ W_ih1^T + h1 @ W_hh1^T + bias ----
    {
      v16h a_x = load_a_frag(&s.h0buf[0][0], LSTM_H, lane, 0);
      v16h a_h = load_a_frag(&s.h1buf[0][0], LSTM_H, lane, 0);
#pragma unroll
      for (int u = 0; u < 2; ++u) {
        int nt = wid * 2 + u;
        float bv = s.bias1[nt * 16 + (lane & 15)];
        v8f acc = {bv, bv, bv, bv, bv, bv, bv, bv};
        acc = wmma_f16(a_x, load_b_frag(&s.wih1[nt][0][0], 32, lane, 0), acc);
        acc = wmma_f16(a_h, load_b_frag(&s.whh1[nt][0][0], 32, lane, 0), acc);
        *(v8f*)&s.gbuf[nt * 16 + (lane & 15)][(lane >> 4) << 3] = acc;
      }
    }
    __syncthreads();  // (D)

    // ---- Layer 1 elementwise ----
    {
      v4f vi = *(const v4f*)&s.gbuf[0 + hh][bb];
      v4f vf = *(const v4f*)&s.gbuf[32 + hh][bb];
      v4f vg = *(const v4f*)&s.gbuf[64 + hh][bb];
      v4f vo = *(const v4f*)&s.gbuf[96 + hh][bb];
#pragma unroll
      for (int j = 0; j < 4; ++j) {
        float ig = fast_sigmoid(vi[j]);
        float fg = fast_sigmoid(vf[j]);
        float gg = fast_tanh(vg[j]);
        float og = fast_sigmoid(vo[j]);
        c1[j] = fg * c1[j] + ig * gg;
        s.h1buf[bb + j][hh] = (_Float16)(og * fast_tanh(c1[j]));
      }
    }
    __syncthreads();  // (E)

    // ---- Output projection as one WMMA on wave 0:
    //      out[b, t, :] = h1(16x32) @ W_out^T(32x16, cols 10..15 zero) + b_out
    if (wid == 0) {
      int n = lane & 15, g = lane >> 4;
      float bv = s.bout16[n];
      v8f acc = {bv, bv, bv, bv, bv, bv, bv, bv};
      v16h a_h = load_a_frag(&s.h1buf[0][0], LSTM_H, lane, 0);
      acc = wmma_f16(a_h, load_b_frag(&s.wout16[0][0], 32, lane, 0), acc);
      if (n < LSTM_OUT) {
#pragma unroll
        for (int v = 0; v < 8; ++v) {
          int m = v + (g << 3);
          out[((size_t)(b0 + m) * LSTM_T + t) * LSTM_OUT + n] = acc[v];
        }
      }
    }
    // next stage-x (xf32) last read before (B); h1buf rewrite fenced by (D).
  }
}

extern "C" void kernel_launch(void* const* d_in, const int* in_sizes, int n_in,
                              void* d_out, int out_size, void* d_ws, size_t ws_size,
                              hipStream_t stream) {
  (void)in_sizes; (void)n_in; (void)out_size; (void)d_ws; (void)ws_size;
  const float* x     = (const float*)d_in[0];
  const float* W_ih0 = (const float*)d_in[1];
  const float* W_hh0 = (const float*)d_in[2];
  const float* b_ih0 = (const float*)d_in[3];
  const float* b_hh0 = (const float*)d_in[4];
  const float* W_ih1 = (const float*)d_in[5];
  const float* W_hh1 = (const float*)d_in[6];
  const float* b_ih1 = (const float*)d_in[7];
  const float* b_hh1 = (const float*)d_in[8];
  const float* W_out = (const float*)d_in[9];
  const float* b_out = (const float*)d_in[10];
  float* out = (float*)d_out;

  dim3 grid(LSTM_B / BTILE);  // 32 workgroups, one 16-batch tile each
  dim3 block(128);            // 4 waves of 32
  lstm2_fused_wmma<<<grid, block, 0, stream>>>(
      x, W_ih0, W_hh0, b_ih0, b_hh0,
      W_ih1, W_hh1, b_ih1, b_hh1, W_out, b_out, out);
}
